// ThresholdedRNNCell_30477087933175
// MI455X (gfx1250) — matrix operand order repaired and checked by reading, not compiled, not run on hardware
//
#include <hip/hip_runtime.h>
#include <hip/hip_bf16.h>
#include <cstdint>

// ---------------------------------------------------------------------------
// ThresholdedRNNCell for MI455X (gfx1250, wave32, WMMA)
//
//   Phase 1:  i2h = x @ W_ih^T + b_ih   -> written straight into d_out
//             (memory-bound; LDS-staged A with in-kernel bf16 split)
//   Phase 2:  for t: h = clamp(relu(i2h[:,t,:] + h @ W_hh^T + b_hh), 0, 1)
//             (latency-bound; hidden state kept in pre-packed bf16 hi/lo
//              WMMA A-operand layout, repacked for free in the epilogue ->
//              inner loop is double-buffered loads + v_wmma only)
//
// All GEMMs: split-precision fp32 = bf16_hi + bf16_lo,
//   acc += Ahi*Bhi + Ahi*Blo + Alo*Bhi   (3x v_wmma_f32_16x16x32_bf16)
// ---------------------------------------------------------------------------

typedef __attribute__((ext_vector_type(16))) __bf16    v16bf;
typedef __attribute__((ext_vector_type(16))) uint16_t  v16u;
typedef __attribute__((ext_vector_type(8)))  float     v8f;

#define BLOCK_THREADS 256      // 8 wave32s per block
#define WAVES_PER_BLOCK 8
#define KSLAB 32               // K per WMMA triple
#define LDS_STRIDE 36          // floats/row: bank-conflict free, 16B aligned

static __device__ __forceinline__ uint16_t f32_to_bf16_rne(float f) {
  uint32_t u = __builtin_bit_cast(uint32_t, f);
  u += 0x7FFFu + ((u >> 16) & 1u);          // round-to-nearest-even
  return (uint16_t)(u >> 16);
}
static __device__ __forceinline__ float bf16_to_f32(uint16_t h) {
  return __builtin_bit_cast(float, ((uint32_t)h) << 16);
}
static __device__ __forceinline__ v8f wmma3_bf16(v8f acc, v16u ah, v16u al,
                                                 v16u bh, v16u bl) {
  const v16bf A_hi = __builtin_bit_cast(v16bf, ah);
  const v16bf A_lo = __builtin_bit_cast(v16bf, al);
  const v16bf B_hi = __builtin_bit_cast(v16bf, bh);
  const v16bf B_lo = __builtin_bit_cast(v16bf, bl);
  acc = __builtin_amdgcn_wmma_f32_16x16x32_bf16(false, A_hi, false, B_hi,
                                                (short)0, acc, false, false);
  acc = __builtin_amdgcn_wmma_f32_16x16x32_bf16(false, A_hi, false, B_lo,
                                                (short)0, acc, false, false);
  acc = __builtin_amdgcn_wmma_f32_16x16x32_bf16(false, A_lo, false, B_hi,
                                                (short)0, acc, false, false);
  return acc;
}

// ---------------------------------------------------------------------------
// B-operand packed layout (bf16): element j of lane L of tile (ks, nt) at
//   ((ks*NT + nt)*32 + L)*16 + j ; n = nt*16 + (L&15), k = ks*32 + (L>=16?8:0) + j + (j>=8?8:0)
// A-operand packed layout (bf16): element j of lane L of tile (mt, ks) at
//   ((mt*KS + ks)*32 + L)*16 + j
//   lanes 0-15 : m = L,    K = {0..7, 16..23}
//   lanes 16-31: m = L-16, K = {8..15, 24..31}
// Each lane's 16 bf16 values are 32 contiguous bytes -> coalesced b128 loads.
// ---------------------------------------------------------------------------

__global__ void pack_w_kernel(const float* __restrict__ W,   // [N,K] row-major
                              uint16_t* __restrict__ hi,
                              uint16_t* __restrict__ lo,
                              int N, int K) {
  const int NT = N >> 4;
  const int total = (K / KSLAB) * NT * 32;
  int tid = blockIdx.x * blockDim.x + threadIdx.x;
  if (tid >= total) return;
  const int lane = tid & 31;
  const int nt   = (tid >> 5) % NT;
  const int ks   = tid / (32 * NT);
  const int n    = nt * 16 + (lane & 15);
  const int k0   = ks * KSLAB + ((lane >> 4) << 3);
  const float* __restrict__ src = W + (size_t)n * K + k0;   // two 32B runs: k0..k0+7, k0+16..k0+23

  uint16_t hb[16], lb[16];
#pragma unroll
  for (int j = 0; j < 16; ++j) {
    float f = src[j < 8 ? j : j + 8];
    uint16_t h = f32_to_bf16_rne(f);
    hb[j] = h;
    lb[j] = f32_to_bf16_rne(f - bf16_to_f32(h));
  }
  const size_t o = (size_t)tid * 16;
  *(uint4*)(hi + o)     = *(uint4*)(hb);
  *(uint4*)(hi + o + 8) = *(uint4*)(hb + 8);
  *(uint4*)(lo + o)     = *(uint4*)(lb);
  *(uint4*)(lo + o + 8) = *(uint4*)(lb + 8);
}

__global__ void pack_a_kernel(const float* __restrict__ A,   // [M,K] row-major
                              uint16_t* __restrict__ hi,
                              uint16_t* __restrict__ lo,
                              int M, int K) {
  const int KS = K >> 5;
  const int total = (M / 16) * KS * 32;
  int tid = blockIdx.x * blockDim.x + threadIdx.x;
  if (tid >= total) return;
  const int lane = tid & 31;
  const int ks   = (tid >> 5) % KS;
  const int mt   = tid / (32 * KS);
  const int m    = mt * 16 + (lane & 15);
  const int kb   = ks * KSLAB + ((lane >> 4) ? 8 : 0);
  const float* __restrict__ src = A + (size_t)m * K + kb;

  uint16_t hb[16], lb[16];
#pragma unroll
  for (int j = 0; j < 8; ++j) {
    float f0 = src[j];        // K group 1
    float f1 = src[16 + j];   // K group 2 (+16)
    uint16_t h0 = f32_to_bf16_rne(f0);
    uint16_t h1 = f32_to_bf16_rne(f1);
    hb[j] = h0;  hb[j + 8] = h1;
    lb[j]     = f32_to_bf16_rne(f0 - bf16_to_f32(h0));
    lb[j + 8] = f32_to_bf16_rne(f1 - bf16_to_f32(h1));
  }
  const size_t o = (size_t)tid * 16;
  *(uint4*)(hi + o)     = *(uint4*)(hb);
  *(uint4*)(hi + o + 8) = *(uint4*)(hb + 8);
  *(uint4*)(lo + o)     = *(uint4*)(lb);
  *(uint4*)(lo + o + 8) = *(uint4*)(lb + 8);
}

// ---------------------------------------------------------------------------
// Phase 1: memory-bound big GEMM. LDS-staged fp32 A tile, split in-kernel.
// Each wave32 produces one 16x16 C tile; 8 waves share the A tile.
// ---------------------------------------------------------------------------
__global__ __launch_bounds__(BLOCK_THREADS)
void i2h_gemm_kernel(const float* __restrict__ A,            // [M,K] fp32
                     const uint16_t* __restrict__ Bhi,
                     const uint16_t* __restrict__ Blo,
                     const float* __restrict__ bias,
                     float* __restrict__ out,                // [M,N]
                     int M, int K, int N) {
  const int lane = threadIdx.x & 31;
  const int wave = threadIdx.x >> 5;
  const int mtile = blockIdx.x;
  const int ntile = blockIdx.y * WAVES_PER_BLOCK + wave;
  const int NT = N >> 4;

  __shared__ float sA[16 * LDS_STRIDE];

  const int row  = lane & 15;
  const int half = lane >> 4;
  const int nslabs = K / KSLAB;
  const size_t bstep = (size_t)NT * 32 * 16;

  v8f acc = {};
  size_t boff = ((size_t)ntile * 32 + lane) * 16;

  for (int ks = 0; ks < nslabs; ++ks) {
    __syncthreads();
    if (threadIdx.x < 128) {   // 16x32 fp32 A tile, 128 x float4 = 2KB
      const int r   = threadIdx.x >> 3;
      const int seg = threadIdx.x & 7;
      const float4 v =
          *(const float4*)(A + (size_t)(mtile * 16 + r) * K + ks * KSLAB + seg * 4);
      *(float4*)(&sA[r * LDS_STRIDE + seg * 4]) = v;
    }
    __syncthreads();

    v16u a_hi, a_lo;
    {
      const float* __restrict__ src = &sA[row * LDS_STRIDE];
      const int kb = half * 8;
#pragma unroll
      for (int j = 0; j < 8; ++j) {
        float f0 = src[kb + j];
        float f1 = src[kb + 16 + j];
        uint16_t h0 = f32_to_bf16_rne(f0);
        uint16_t h1 = f32_to_bf16_rne(f1);
        a_hi[j] = h0;  a_hi[j + 8] = h1;
        a_lo[j]     = f32_to_bf16_rne(f0 - bf16_to_f32(h0));
        a_lo[j + 8] = f32_to_bf16_rne(f1 - bf16_to_f32(h1));
      }
    }

    v16u b_hi = *(const v16u*)(Bhi + boff);
    v16u b_lo = *(const v16u*)(Blo + boff);
    if (ks + 1 < nslabs) {
      __builtin_prefetch(Bhi + boff + bstep, 0, 0);   // global_prefetch_b8
      __builtin_prefetch(Blo + boff + bstep, 0, 0);
    }
    boff += bstep;

    acc = wmma3_bf16(acc, a_hi, a_lo, b_hi, b_lo);
  }

  // C layout: VGPR r, lane L -> M = r + 8*(L>>4), Nc = L&15.
  const int col = ntile * 16 + row;
  const float bv = bias[col];
#pragma unroll
  for (int r = 0; r < 8; ++r) {
    const int mrow = mtile * 16 + r + 8 * half;
    out[(size_t)mrow * N + col] = acc[r] + bv;
  }
}

// ---------------------------------------------------------------------------
// Phase 2: one serialized RNN step. Software-pipelined: next slab's four
// 32B fragment loads are issued before the current slab's WMMAs, so load
// latency overlaps matrix issue. Epilogue clamps, writes rnn_out, and
// repacks h into the NEXT step's packed A buffer (each element owned once).
// ---------------------------------------------------------------------------
__global__ __launch_bounds__(BLOCK_THREADS)
void rnn_step_kernel(const uint16_t* __restrict__ Ahi,
                     const uint16_t* __restrict__ Alo,
                     const uint16_t* __restrict__ Bhi,
                     const uint16_t* __restrict__ Blo,
                     const float* __restrict__ bias,
                     float* __restrict__ out,                // rnn_out [B,T,N]
                     uint16_t* __restrict__ nAhi,            // next packed h
                     uint16_t* __restrict__ nAlo,
                     int K, int N, int T, int t) {
  const int lane = threadIdx.x & 31;
  const int wave = threadIdx.x >> 5;
  const int mtile = blockIdx.x;
  const int ntile = blockIdx.y * WAVES_PER_BLOCK + wave;
  const int NT = N >> 4;
  const int KS = K >> 5;

  v8f acc = {};
  size_t aoff = ((size_t)(mtile * KS) * 32 + lane) * 16;
  size_t boff = ((size_t)ntile * 32 + lane) * 16;
  const size_t bstep = (size_t)NT * 32 * 16;

  // Prologue: fragments for slab 0.
  v16u a_hi = *(const v16u*)(Ahi + aoff);
  v16u a_lo = *(const v16u*)(Alo + aoff);
  v16u b_hi = *(const v16u*)(Bhi + boff);
  v16u b_lo = *(const v16u*)(Blo + boff);

#pragma unroll 2
  for (int ks = 0; ks < KS - 1; ++ks) {
    aoff += 32 * 16;
    boff += bstep;
    // Issue next slab's loads BEFORE computing on current fragments.
    v16u a_hi_n = *(const v16u*)(Ahi + aoff);
    v16u a_lo_n = *(const v16u*)(Alo + aoff);
    v16u b_hi_n = *(const v16u*)(Bhi + boff);
    v16u b_lo_n = *(const v16u*)(Blo + boff);
    __builtin_prefetch(Bhi + boff + bstep, 0, 0);   // global_prefetch_b8
    __builtin_prefetch(Blo + boff + bstep, 0, 0);

    acc = wmma3_bf16(acc, a_hi, a_lo, b_hi, b_lo);

    a_hi = a_hi_n;  a_lo = a_lo_n;
    b_hi = b_hi_n;  b_lo = b_lo_n;
  }
  acc = wmma3_bf16(acc, a_hi, a_lo, b_hi, b_lo);    // drain

  const int row  = lane & 15;
  const int half = lane >> 4;
  const int col  = ntile * 16 + row;
  const float bv = bias[col];

  // Packed-A coordinates for h element (m, k=col) of the NEXT step (K'=N):
  const int KSn  = N >> 5;
  const int ks2  = col >> 5;
  const int kk   = col & 31;
  const int g    = kk >> 3;                       // k-group 0..3
  const int j2   = (kk & 7) + ((g >> 1) << 3);
  const int l2b  = (g & 1) << 4;                  // lane half select

#pragma unroll
  for (int r = 0; r < 8; ++r) {
    const int mrow = mtile * 16 + r + 8 * half;   // m in [0,B)
    const size_t o = ((size_t)mrow * T + t) * (size_t)N + col;
    float pre = out[o] + acc[r] + bv;
    float h = fminf(fmaxf(pre, 0.0f), 1.0f);      // min(relu(pre), 1.0)
    out[o] = h;
    const uint16_t hb = f32_to_bf16_rne(h);
    const uint16_t lb = f32_to_bf16_rne(h - bf16_to_f32(hb));
    const size_t p =
        ((size_t)(mtile * KSn + ks2) * 32 + (r + 8 * half + l2b)) * 16 + j2;
    nAhi[p] = hb;
    nAlo[p] = lb;
  }
}

// h_final == rnn_out[:, T-1, :]; gather into tail of d_out.
__global__ void tail_kernel(float* __restrict__ out, int B, int T, int H) {
  int i = blockIdx.x * blockDim.x + threadIdx.x;
  if (i >= B * H) return;
  const int b = i / H, h = i - b * H;
  out[(size_t)B * T * H + i] = out[((size_t)b * T + (T - 1)) * (size_t)H + h];
}

// ---------------------------------------------------------------------------
extern "C" void kernel_launch(void* const* d_in, const int* in_sizes, int n_in,
                              void* d_out, int out_size, void* d_ws, size_t ws_size,
                              hipStream_t stream) {
  const float* x    = (const float*)d_in[0];   // [B,T,I]
  const float* h0   = (const float*)d_in[1];   // [B,H]
  const float* W_ih = (const float*)d_in[2];   // [H,I]
  const float* b_ih = (const float*)d_in[3];   // [H]
  const float* W_hh = (const float*)d_in[4];   // [H,H]
  const float* b_hh = (const float*)d_in[5];   // [H]
  float* out = (float*)d_out;                  // rnn_out [B,T,H] ++ h_final [B,H]

  const int B = 256, T = 512, I = 512, H = 1024;

  // Workspace (~8 MB): packed weights + ping-pong packed hidden state.
  uint16_t* wih_hi = (uint16_t*)d_ws;
  uint16_t* wih_lo = wih_hi + (size_t)I * H;
  uint16_t* whh_hi = wih_lo + (size_t)I * H;
  uint16_t* whh_lo = whh_hi + (size_t)H * H;
  uint16_t* hA_hi[2], *hA_lo[2];
  hA_hi[0] = whh_lo + (size_t)H * H;
  hA_lo[0] = hA_hi[0] + (size_t)B * H;
  hA_hi[1] = hA_lo[0] + (size_t)B * H;
  hA_lo[1] = hA_hi[1] + (size_t)B * H;

  // 1) Pack weights (bf16 hi/lo, WMMA B-operand layout).
  {
    int tot_ih = (I / KSLAB) * (H / 16) * 32;
    pack_w_kernel<<<(tot_ih + 255) / 256, 256, 0, stream>>>(W_ih, wih_hi, wih_lo, H, I);
    int tot_hh = (H / KSLAB) * (H / 16) * 32;
    pack_w_kernel<<<(tot_hh + 255) / 256, 256, 0, stream>>>(W_hh, whh_hi, whh_lo, H, H);
  }

  // 2) Pack initial hidden state into WMMA A-operand layout.
  {
    int tot = (B / 16) * (H / KSLAB) * 32;
    pack_a_kernel<<<(tot + 255) / 256, 256, 0, stream>>>(h0, hA_hi[0], hA_lo[0], B, H);
  }

  // 3) Phase 1: i2h + b_ih -> d_out (memory-bound).
  {
    const int M = B * T;
    dim3 grid(M / 16, (H / 16) / WAVES_PER_BLOCK);
    i2h_gemm_kernel<<<grid, BLOCK_THREADS, 0, stream>>>(
        x, wih_hi, wih_lo, b_ih, out, M, I, H);
  }

  // 4) Phase 2: 512 serialized steps; W_hh packed (4MB) stays in L2.
  {
    dim3 grid(B / 16, (H / 16) / WAVES_PER_BLOCK);
    for (int t = 0; t < T; ++t) {
      const int cur = t & 1, nxt = cur ^ 1;
      rnn_step_kernel<<<grid, BLOCK_THREADS, 0, stream>>>(
          hA_hi[cur], hA_lo[cur], whh_hi, whh_lo, b_hh, out,
          hA_hi[nxt], hA_lo[nxt], H, H, T, t);
    }
  }

  // 5) h_final = rnn_out[:, T-1, :] -> tail of d_out.
  tail_kernel<<<(B * H + 255) / 256, 256, 0, stream>>>(out, B, T, H);
}
